// CrossAttention2d_1898375545163
// MI455X (gfx1250) — matrix operand order repaired and checked by reading, not compiled, not run on hardware
//
#include <hip/hip_runtime.h>

// ---------------------------------------------------------------------------
// CrossAttention2d for MI455X (gfx1250, wave32, WMMA f32_16x16x32_f16)
// N=16, C_DEC=512, H=W=32 (HW=1024), C_ENC=768, S=77, NHEAD=8, D=64
// Big projections use 2x4 (32x64) macro-tiles per wave for register reuse:
// 12 b128 fragment loads per 8 WMMAs instead of 4 loads per WMMA.
// ---------------------------------------------------------------------------

#define NB    16
#define CDEC  512
#define HW    1024
#define CENC  768
#define SEQ   77
#define SPAD  80     // S padded to 5 x 16 tiles
#define SPAD2 96     // S padded to 3 x 32 K-chunks for the PV GEMM
#define NHEAD 8
#define DHEAD 64

typedef __attribute__((ext_vector_type(16))) _Float16 v16h;
typedef __attribute__((ext_vector_type(8)))  _Float16 v8h;
typedef __attribute__((ext_vector_type(8)))  float    v8f;

// A fragment (16 rows, K=32), per ISA 16-bit A layout:
// lane L: row = L&15; halves 0..7 -> K = hs*8.. ; halves 8..15 -> K = 16+hs*8..
__device__ __forceinline__ v16h load_a_frag(const _Float16* tile, int ld, int lane) {
  int r  = lane & 15;
  int hs = (lane >> 4) & 1;
  const _Float16* row = tile + (size_t)r * ld;
  v8h lo = *(const v8h*)(row + hs * 8);
  v8h hi = *(const v8h*)(row + 16 + hs * 8);
  v16h a;
#pragma unroll
  for (int i = 0; i < 8; ++i) { a[i] = lo[i]; a[8 + i] = hi[i]; }
  return a;
}

// B fragment from a B^T ([N][K] row-major) buffer, K=32:
// lane L: col = L&15; halves 0..7 -> K = hs*8.. ; halves 8..15 -> K = 16+hs*8..
__device__ __forceinline__ v16h load_b_frag(const _Float16* btile, int ld, int lane) {
  int c  = lane & 15;
  int hs = (lane >> 4) & 1;
  const _Float16* p = btile + (size_t)c * ld + hs * 8;
  v8h lo = *(const v8h*)(p);
  v8h hi = *(const v8h*)(p + 16);
  v16h b;
#pragma unroll
  for (int i = 0; i < 8; ++i) { b[i] = lo[i]; b[8 + i] = hi[i]; }
  return b;
}

__device__ __forceinline__ v8f wmma16(v16h a, v16h b, v8f c) {
  return __builtin_amdgcn_wmma_f32_16x16x32_f16(false, a, false, b, (short)0, c, false, false);
}

// 2x4 macro-tile GEMM core: C[32 x 64] += A[32 x K] * B^T[64 x K]^T
__device__ __forceinline__ void gemm_core_2x4(const _Float16* __restrict__ A0,
                                              const _Float16* __restrict__ B0,
                                              int lda, int ldb, int K, int lane,
                                              v8f acc[2][4]) {
  for (int kk = 0; kk < K; kk += 32) {
    v16h a0 = load_a_frag(A0 + kk, lda, lane);
    v16h a1 = load_a_frag(A0 + (size_t)16 * lda + kk, lda, lane);
    v16h b0 = load_b_frag(B0 + kk, ldb, lane);
    v16h b1 = load_b_frag(B0 + (size_t)16 * ldb + kk, ldb, lane);
    v16h b2 = load_b_frag(B0 + (size_t)32 * ldb + kk, ldb, lane);
    v16h b3 = load_b_frag(B0 + (size_t)48 * ldb + kk, ldb, lane);
    acc[0][0] = wmma16(a0, b0, acc[0][0]);
    acc[0][1] = wmma16(a0, b1, acc[0][1]);
    acc[0][2] = wmma16(a0, b2, acc[0][2]);
    acc[0][3] = wmma16(a0, b3, acc[0][3]);
    acc[1][0] = wmma16(a1, b0, acc[1][0]);
    acc[1][1] = wmma16(a1, b1, acc[1][1]);
    acc[1][2] = wmma16(a1, b2, acc[1][2]);
    acc[1][3] = wmma16(a1, b3, acc[1][3]);
  }
}

// ---------------------------------------------------------------------------
// 1) per-sample mean / rstd over (C,H,W)
__global__ void stats_kernel(const float* __restrict__ input, float* __restrict__ stats) {
  int n = blockIdx.x;
  const float* p = input + (size_t)n * CDEC * HW;
  float s = 0.f, s2 = 0.f;
  for (int i = threadIdx.x; i < CDEC * HW; i += 256) {
    float v = p[i]; s += v; s2 += v * v;
  }
  __shared__ float red[512];
  red[threadIdx.x] = s; red[256 + threadIdx.x] = s2;
  __syncthreads();
  for (int off = 128; off > 0; off >>= 1) {
    if (threadIdx.x < off) {
      red[threadIdx.x] += red[threadIdx.x + off];
      red[256 + threadIdx.x] += red[256 + threadIdx.x + off];
    }
    __syncthreads();
  }
  if (threadIdx.x == 0) {
    float inv = 1.f / (CDEC * HW);
    float mu  = red[0] * inv;
    float var = red[256] * inv - mu * mu;
    stats[2 * n] = mu;
    stats[2 * n + 1] = rsqrtf(var + 1e-5f);
  }
}

// 2) ss[n, 0:1024] = cond @ adagn_w^T + adagn_b  (tiny)
__global__ void adagn_kernel(const float* __restrict__ cond, const float* __restrict__ w,
                             const float* __restrict__ b, float* __restrict__ ss) {
  int gid = blockIdx.x * blockDim.x + threadIdx.x;   // 16*1024
  int n = gid >> 10, j = gid & 1023;
  const float* cn = cond + (size_t)n * 512;
  const float* wj = w + (size_t)j * 512;
  float acc = b[j];
  for (int k = 0; k < 512; ++k) acc += cn[k] * wj[k];
  ss[gid] = acc;
}

// 3a) f32 -> f16 conversion / zero fill
__global__ void cvt_kernel(const float* __restrict__ src, _Float16* __restrict__ dst, int n) {
  for (int i = blockIdx.x * blockDim.x + threadIdx.x; i < n; i += gridDim.x * blockDim.x)
    dst[i] = (_Float16)src[i];
}
__global__ void zero_h_kernel(_Float16* __restrict__ dst, int n) {
  for (int i = blockIdx.x * blockDim.x + threadIdx.x; i < n; i += gridDim.x * blockDim.x)
    dst[i] = (_Float16)0.f;
}

// 3b) LayerNorm of enc_hidden -> e[n][SPAD][768] f16, zero rows s>=77
__global__ void ln_kernel(const float* __restrict__ eh, const float* __restrict__ lnw,
                          const float* __restrict__ lnb, _Float16* __restrict__ e) {
  int bid = blockIdx.x;
  int n = bid / SPAD, s = bid % SPAD;
  _Float16* dst = e + ((size_t)n * SPAD + s) * CENC;
  if (s >= SEQ) {
    for (int i = threadIdx.x; i < CENC; i += 128) dst[i] = (_Float16)0.f;
    return;
  }
  const float* src = eh + ((size_t)n * SEQ + s) * CENC;
  float sum = 0.f, s2 = 0.f;
  for (int i = threadIdx.x; i < CENC; i += 128) { float v = src[i]; sum += v; s2 += v * v; }
  __shared__ float red[256];
  red[threadIdx.x] = sum; red[128 + threadIdx.x] = s2;
  __syncthreads();
  for (int off = 64; off > 0; off >>= 1) {
    if (threadIdx.x < off) {
      red[threadIdx.x] += red[threadIdx.x + off];
      red[128 + threadIdx.x] += red[128 + threadIdx.x + off];
    }
    __syncthreads();
  }
  float mu = red[0] / CENC;
  float rstd = rsqrtf(red[128] / CENC - mu * mu + 1e-5f);
  for (int i = threadIdx.x; i < CENC; i += 128)
    dst[i] = (_Float16)((src[i] - mu) * rstd * lnw[i] + lnb[i]);
}

// 4) modulate + transpose: Xt[n][hw][c] = shift + normed*(scale+1), f16
__global__ void modulate_kernel(const float* __restrict__ input, const float* __restrict__ stats,
                                const float* __restrict__ ss, _Float16* __restrict__ xt) {
  __shared__ float tile[32][33];
  int n = blockIdx.z, hwT = blockIdx.x, cT = blockIdx.y;
  int tx = threadIdx.x, ty = threadIdx.y;          // (32, 8)
  float mu = stats[2 * n], rstd = stats[2 * n + 1];
  const float* ssn = ss + (size_t)n * 1024;
#pragma unroll
  for (int i = 0; i < 4; ++i) {
    int c = cT * 32 + ty + i * 8;
    int hw = hwT * 32 + tx;
    float v = input[((size_t)n * CDEC + c) * HW + hw];
    tile[ty + i * 8][tx] = (v - mu) * rstd * (1.f + ssn[c]) + ssn[512 + c];
  }
  __syncthreads();
#pragma unroll
  for (int i = 0; i < 4; ++i) {
    int hw = hwT * 32 + ty + i * 8;
    int c = cT * 32 + tx;
    xt[((size_t)n * HW + hw) * CDEC + c] = (_Float16)tile[tx][ty + i * 8];
  }
}

// 5) Q = x @ q_w^T. C=[hw x o], 32x64 per wave. Stores Qt[n][hw][o]*d^-0.25 f16.
__global__ void gemm_q_kernel(const _Float16* __restrict__ xt, const _Float16* __restrict__ wq,
                              const float* __restrict__ qb, _Float16* __restrict__ qt) {
  int wid = (blockIdx.x * blockDim.x + threadIdx.x) >> 5;
  int lane = threadIdx.x & 31;
  int n = wid >> 8;                 // 32 mtiles * 8 ntiles = 256 macro-tiles per n
  int t = wid & 255;
  int mt = t >> 3, nt = t & 7;      // mt: 32-row strip, nt: 64-col strip
  const _Float16* A0 = xt + ((size_t)n * HW + mt * 32) * CDEC;
  const _Float16* B0 = wq + (size_t)(nt * 64) * CDEC;
  v8f acc[2][4] = {};
  gemm_core_2x4(A0, B0, CDEC, CDEC, CDEC, lane, acc);
  int l15 = lane & 15, hs = (lane >> 4) & 1;
  const float DS = 0.35355339059f;  // 64^-0.25
#pragma unroll
  for (int ni = 0; ni < 4; ++ni) {
    int col = nt * 64 + ni * 16 + l15;
    float bias = qb[col];
#pragma unroll
    for (int mi = 0; mi < 2; ++mi)
#pragma unroll
      for (int v = 0; v < 8; ++v) {
        int row = mt * 32 + mi * 16 + v + 8 * hs;
        qt[((size_t)n * HW + row) * CDEC + col] = (_Float16)((acc[mi][ni][v] + bias) * DS);
      }
  }
}

// 6) KV = e @ kv_w^T, C=[o x s]; scatter K->[n,h,s,d]*d^-0.25, V->[n,h,d,SPAD2]
__global__ void gemm_kv_kernel(const _Float16* __restrict__ e, const _Float16* __restrict__ wkv,
                               const float* __restrict__ kvb, _Float16* __restrict__ kmat,
                               _Float16* __restrict__ vt) {
  int wid = (blockIdx.x * blockDim.x + threadIdx.x) >> 5;
  int lane = threadIdx.x & 31;
  int n = wid / 320;                // 64 ot * 5 st = 320 tiles per n
  int t = wid % 320;
  int ot = t / 5, st = t % 5;
  const _Float16* A0 = wkv + (size_t)(ot * 16) * CENC;
  const _Float16* B0 = e + ((size_t)n * SPAD + st * 16) * CENC;
  v8f acc = {};
  for (int kk = 0; kk < CENC; kk += 32)
    acc = wmma16(load_a_frag(A0 + kk, CENC, lane), load_b_frag(B0 + kk, CENC, lane), acc);
  int s = st * 16 + (lane & 15);
  int hs = (lane >> 4) & 1;
  const float DS = 0.35355339059f;
#pragma unroll
  for (int v = 0; v < 8; ++v) {
    int o = ot * 16 + v + 8 * hs;
    float val = acc[v] + kvb[o];
    if (o < 512) {
      int head = o >> 6, dd = o & 63;
      kmat[(((size_t)n * NHEAD + head) * SPAD + s) * DHEAD + dd] = (_Float16)(val * DS);
    } else {
      int o2 = o - 512, head = o2 >> 6, dd = o2 & 63;
      vt[(((size_t)n * NHEAD + head) * DHEAD + dd) * SPAD2 + s] = (_Float16)val;
    }
  }
}

// 7) attention: per (n,head) block, 4 waves x 16 q-tiles each
__global__ void attn_kernel(const _Float16* __restrict__ qt, const _Float16* __restrict__ kmat,
                            const _Float16* __restrict__ vt, const float* __restrict__ mask,
                            _Float16* __restrict__ yt) {
  int n = blockIdx.x >> 3, head = blockIdx.x & 7;
  int wave = threadIdx.x >> 5, lane = threadIdx.x & 31;
  int l15 = lane & 15, hs = (lane >> 4) & 1;
  __shared__ _Float16 P[4][16 * SPAD2];
  _Float16* Pw = P[wave];
  const _Float16* Kb = kmat + ((size_t)n * NHEAD + head) * SPAD * DHEAD;
  const _Float16* Vb = vt + ((size_t)n * NHEAD + head) * DHEAD * SPAD2;
  const float* mk = mask + (size_t)n * SEQ;

  for (int qtile = wave; qtile < 64; qtile += 4) {
    const _Float16* Qb = qt + ((size_t)n * HW + qtile * 16) * CDEC + head * DHEAD;
    v8f logit[5];
#pragma unroll
    for (int st = 0; st < 5; ++st) {
      v8f acc = {};
#pragma unroll
      for (int kk = 0; kk < DHEAD; kk += 32)
        acc = wmma16(load_a_frag(Qb + kk, CDEC, lane),
                     load_b_frag(Kb + (size_t)(st * 16) * DHEAD + kk, DHEAD, lane), acc);
      int s = st * 16 + l15;
      float madd = (s < SEQ) ? (-10000.f * mk[s]) : -1e30f;
#pragma unroll
      for (int v = 0; v < 8; ++v) acc[v] += madd;
      logit[st] = acc;
    }
    // softmax across 80 columns of each of the 8 rows this lane holds
    float ex[5][8];
#pragma unroll
    for (int v = 0; v < 8; ++v) {
      float m = logit[0][v];
#pragma unroll
      for (int st = 1; st < 5; ++st) m = fmaxf(m, logit[st][v]);
      m = fmaxf(m, __shfl_xor(m, 1, 32));
      m = fmaxf(m, __shfl_xor(m, 2, 32));
      m = fmaxf(m, __shfl_xor(m, 4, 32));
      m = fmaxf(m, __shfl_xor(m, 8, 32));
      float sum = 0.f;
#pragma unroll
      for (int st = 0; st < 5; ++st) { ex[st][v] = __expf(logit[st][v] - m); sum += ex[st][v]; }
      sum += __shfl_xor(sum, 1, 32);
      sum += __shfl_xor(sum, 2, 32);
      sum += __shfl_xor(sum, 4, 32);
      sum += __shfl_xor(sum, 8, 32);
      float rinv = 1.f / sum;
#pragma unroll
      for (int st = 0; st < 5; ++st) ex[st][v] *= rinv;
    }
    // stage P into LDS as next A operand (rows 16 x SPAD2, zero s>=80)
#pragma unroll
    for (int st = 0; st < 5; ++st)
#pragma unroll
      for (int v = 0; v < 8; ++v)
        Pw[(v + 8 * hs) * SPAD2 + st * 16 + l15] = (_Float16)ex[st][v];
#pragma unroll
    for (int v = 0; v < 8; ++v)
      Pw[(v + 8 * hs) * SPAD2 + 80 + l15] = (_Float16)0.f;
    // y = P @ V
#pragma unroll
    for (int dt = 0; dt < 4; ++dt) {
      v8f acc = {};
#pragma unroll
      for (int kk = 0; kk < SPAD2; kk += 32)
        acc = wmma16(load_a_frag(Pw + kk, SPAD2, lane),
                     load_b_frag(Vb + (size_t)(dt * 16) * SPAD2 + kk, SPAD2, lane), acc);
      int o = head * DHEAD + dt * 16 + l15;
#pragma unroll
      for (int v = 0; v < 8; ++v) {
        int hw = qtile * 16 + v + 8 * hs;
        yt[((size_t)n * HW + hw) * CDEC + o] = (_Float16)acc[v];
      }
    }
  }
}

// 8) out = y @ out_w^T + out_b + input. C=[o x hw], 32x64 per wave, fp32 store.
__global__ void gemm_out_kernel(const _Float16* __restrict__ yt, const _Float16* __restrict__ wout,
                                const float* __restrict__ ob, const float* __restrict__ input,
                                float* __restrict__ out) {
  int wid = (blockIdx.x * blockDim.x + threadIdx.x) >> 5;
  int lane = threadIdx.x & 31;
  int n = wid >> 8;                 // 16 mtiles * 16 ntiles = 256 macro-tiles per n
  int t = wid & 255;
  int mt = t >> 4, nt = t & 15;     // mt: 32 o-rows, nt: 64 hw-cols
  const _Float16* A0 = wout + (size_t)(mt * 32) * CDEC;
  const _Float16* B0 = yt + ((size_t)n * HW + nt * 64) * CDEC;
  v8f acc[2][4] = {};
  gemm_core_2x4(A0, B0, CDEC, CDEC, CDEC, lane, acc);
  int l15 = lane & 15, hs = (lane >> 4) & 1;
#pragma unroll
  for (int mi = 0; mi < 2; ++mi)
#pragma unroll
    for (int v = 0; v < 8; ++v) {
      int o = mt * 32 + mi * 16 + v + 8 * hs;
      float bias = ob[o];
#pragma unroll
      for (int ni = 0; ni < 4; ++ni) {
        int hw = nt * 64 + ni * 16 + l15;
        size_t idx = ((size_t)n * CDEC + o) * HW + hw;
        out[idx] = input[idx] + acc[mi][ni][v] + bias;
      }
    }
}

// ---------------------------------------------------------------------------
extern "C" void kernel_launch(void* const* d_in, const int* in_sizes, int n_in,
                              void* d_out, int out_size, void* d_ws, size_t ws_size,
                              hipStream_t stream) {
  (void)in_sizes; (void)n_in; (void)out_size; (void)ws_size;
  const float* in_input = (const float*)d_in[0];
  const float* in_cond  = (const float*)d_in[1];
  const float* in_enc   = (const float*)d_in[2];
  const float* in_mask  = (const float*)d_in[3];
  const float* in_agw   = (const float*)d_in[4];
  const float* in_agb   = (const float*)d_in[5];
  const float* in_lnw   = (const float*)d_in[6];
  const float* in_lnb   = (const float*)d_in[7];
  const float* in_qw    = (const float*)d_in[8];
  const float* in_qb    = (const float*)d_in[9];
  const float* in_kvw   = (const float*)d_in[10];
  const float* in_kvb   = (const float*)d_in[11];
  const float* in_outw  = (const float*)d_in[12];
  const float* in_outb  = (const float*)d_in[13];
  float* out = (float*)d_out;

  char* ws = (char*)d_ws;
  size_t off = 0;
  auto alloc = [&](size_t bytes) { size_t o = off; off += (bytes + 255) & ~(size_t)255; return o; };
  float*    stats = (float*)(ws + alloc((size_t)NB * 2 * 4));
  float*    ss    = (float*)(ws + alloc((size_t)NB * 1024 * 4));
  _Float16* wq    = (_Float16*)(ws + alloc((size_t)CDEC * CDEC * 2));
  _Float16* wkv   = (_Float16*)(ws + alloc((size_t)1024 * CENC * 2));
  _Float16* wout  = (_Float16*)(ws + alloc((size_t)CDEC * CDEC * 2));
  _Float16* e     = (_Float16*)(ws + alloc((size_t)NB * SPAD * CENC * 2));
  _Float16* xt    = (_Float16*)(ws + alloc((size_t)NB * HW * CDEC * 2));
  _Float16* qt    = (_Float16*)(ws + alloc((size_t)NB * HW * CDEC * 2));
  _Float16* kmat  = (_Float16*)(ws + alloc((size_t)NB * NHEAD * SPAD * DHEAD * 2));
  _Float16* vt    = (_Float16*)(ws + alloc((size_t)NB * NHEAD * DHEAD * SPAD2 * 2));
  _Float16* yt    = (_Float16*)(ws + alloc((size_t)NB * HW * CDEC * 2));

  stats_kernel<<<NB, 256, 0, stream>>>(in_input, stats);
  adagn_kernel<<<64, 256, 0, stream>>>(in_cond, in_agw, in_agb, ss);
  cvt_kernel<<<256, 256, 0, stream>>>(in_qw,  wq,  CDEC * CDEC);
  cvt_kernel<<<256, 256, 0, stream>>>(in_kvw, wkv, 1024 * CENC);
  cvt_kernel<<<256, 256, 0, stream>>>(in_outw, wout, CDEC * CDEC);
  zero_h_kernel<<<256, 256, 0, stream>>>(vt, NB * NHEAD * DHEAD * SPAD2);
  ln_kernel<<<NB * SPAD, 128, 0, stream>>>(in_enc, in_lnw, in_lnb, e);
  modulate_kernel<<<dim3(32, 16, NB), dim3(32, 8), 0, stream>>>(in_input, stats, ss, xt);
  gemm_q_kernel<<<512, 256, 0, stream>>>(xt, wq, in_qb, qt);
  gemm_kv_kernel<<<640, 256, 0, stream>>>(e, wkv, in_kvb, kmat, vt);
  attn_kernel<<<NB * NHEAD, 128, 0, stream>>>(qt, kmat, vt, in_mask, yt);
  gemm_out_kernel<<<512, 256, 0, stream>>>(yt, wout, in_outb, in_input, out);
}
